// DeepLabHeadV3Plus_32177894982021
// MI455X (gfx1250) — matrix operand order repaired and checked by reading, not compiled, not run on hardware
//
#include <hip/hip_runtime.h>
#include <hip/hip_bf16.h>
#include <stdint.h>

typedef uint16_t u16;
typedef __attribute__((ext_vector_type(8)))  uint16_t u16x8;
typedef __attribute__((ext_vector_type(16))) __bf16   v16bf;
typedef __attribute__((ext_vector_type(8)))  float    v8f;

union FragU { u16x8 h[2]; v16bf v; };

__device__ __forceinline__ u16 f2bf(float x) {
  uint32_t u = __float_as_uint(x);
  u += 0x7FFFu + ((u >> 16) & 1u);
  return (u16)(u >> 16);
}
__device__ __forceinline__ float bf2f(u16 h) {
  return __uint_as_float(((uint32_t)h) << 16);
}

// ---------------- layout conversion kernels ----------------

// NCHW f32 -> NHWC bf16 (channel dim padded to Cpad with zeros)
__global__ void k_cast_nhwc(const float* __restrict__ src, u16* __restrict__ dst,
                            int N, int C, int HW, int Cpad) {
  const long long total = (long long)N * HW * Cpad;
  for (long long i = (long long)blockIdx.x * blockDim.x + threadIdx.x; i < total;
       i += (long long)gridDim.x * blockDim.x) {
    int c = (int)(i % Cpad);
    long long row = i / Cpad;
    int p = (int)(row % HW);
    int n = (int)(row / HW);
    dst[i] = (c < C) ? f2bf(src[((long long)n * C + c) * HW + p]) : (u16)0;
  }
}

// OIHW f32 weights -> bf16 [taps][Copad][Kpad] (zero padded)
__global__ void k_wcast(const float* __restrict__ w, u16* __restrict__ dst,
                        int taps, int Co, int Kin, int Copad, int Kpad) {
  const long long total = (long long)taps * Copad * Kpad;
  for (long long i = (long long)blockIdx.x * blockDim.x + threadIdx.x; i < total;
       i += (long long)gridDim.x * blockDim.x) {
    int k = (int)(i % Kpad);
    long long r = i / Kpad;
    int co = (int)(r % Copad);
    int t = (int)(r / Copad);
    u16 v = 0;
    if (co < Co && k < Kin)
      v = f2bf(w[((long long)co * Kin + k) * taps + t]);
    dst[i] = v;
  }
}

// ---------------- WMMA implicit-GEMM conv kernel ----------------
// X: NHWC bf16 [N*H*W][Kpad], Wt: bf16 [taps][Copad][Kpad], Out: f32 [N*H*W][Copad]
// Zrow: >= Kpad zero bf16 values (zero-page for out-of-bounds taps).
// Each wave computes a 32(co) x 64(m) register tile (2 A-frags x 4 B-frags,
// 8 WMMAs per K-step) with double-buffered K (Kpad % 64 == 0, Kpad >= 128).

#define LOAD_A(f, ptr, kb) { (f).h[0] = *(const u16x8*)((ptr) + (kb));      \
                             (f).h[1] = *(const u16x8*)((ptr) + (kb) + 16); }
#define LOAD_B(f, ptr, kb) { (f).h[0] = *(const u16x8*)((ptr) + (kb));      \
                             (f).h[1] = *(const u16x8*)((ptr) + (kb) + 16); }
#define WMMA(acc, A, B) (acc) = __builtin_amdgcn_wmma_f32_16x16x32_bf16(    \
    false, (A).v, false, (B).v, (short)0, (acc), false, false)
#define WMMA_GROUP(fa0, fa1, fb0, fb1, fb2, fb3)                            \
  { WMMA(acc[0][0], fa0, fb0); WMMA(acc[0][1], fa0, fb1);                   \
    WMMA(acc[0][2], fa0, fb2); WMMA(acc[0][3], fa0, fb3);                   \
    WMMA(acc[1][0], fa1, fb0); WMMA(acc[1][1], fa1, fb1);                   \
    WMMA(acc[1][2], fa1, fb2); WMMA(acc[1][3], fa1, fb3); }
#define LOAD_ALL(fa0, fa1, fb0, fb1, fb2, fb3, kk)                          \
  { LOAD_A(fa0, ap0, kk);   LOAD_A(fa1, ap1, kk);                           \
    LOAD_B(fb0, bp[0], kk); LOAD_B(fb1, bp[1], kk);                         \
    LOAD_B(fb2, bp[2], kk); LOAD_B(fb3, bp[3], kk); }

__global__ void __launch_bounds__(256, 2)
k_gemm_wmma(const u16* __restrict__ X, const u16* __restrict__ Wt,
            const u16* __restrict__ Zrow,
            float* __restrict__ Out,
            int N, int H, int W, int Kpad, int Copad,
            int taps, int dil) {
  const int lane = threadIdx.x & 31;
  const int laneRow = lane & 15;
  const int laneHalf = lane >> 4;
  const int tilesCo = Copad >> 5;
  const int M = N * H * W;
  const int tilesM = M >> 6;
  const int wave = (int)(((long long)blockIdx.x * blockDim.x + threadIdx.x) >> 5);
  if (wave >= tilesCo * tilesM) return;  // wave-uniform exit

  const int coT = wave % tilesCo;
  const int mT  = wave / tilesCo;
  const int m0  = mT << 6;
  const int n   = m0 / (H * W);
  const int rem = m0 - n * H * W;
  const int h   = rem / W;
  const int w0  = rem - h * W;

  const v8f zf = {0.f, 0.f, 0.f, 0.f, 0.f, 0.f, 0.f, 0.f};
  v8f acc[2][4];
#pragma unroll
  for (int i = 0; i < 2; ++i)
#pragma unroll
    for (int j = 0; j < 4; ++j) acc[i][j] = zf;

  for (int tap = 0; tap < taps; ++tap) {
    int dh = 0, dw = 0;
    if (taps == 9) { dh = (tap / 3 - 1) * dil; dw = (tap % 3 - 1) * dil; }
    const int hh = h + dh;
    if ((unsigned)hh >= (unsigned)H) continue;  // uniform per tile

    // A fragment (16x32 bf16): lane holds row co; K halves {lh*8..+7},{16+lh*8..+7}
    const u16* Wtap = Wt + (size_t)tap * Copad * Kpad;
    const u16* ap0 = Wtap + (size_t)((coT << 5) + laneRow) * Kpad + laneHalf * 8;
    const u16* ap1 = ap0 + (size_t)16 * Kpad;

    // B fragments (32x16 bf16): lane holds column m; K halves {lh*8..+7},{16+lh*8..+7}
    // OOB lanes are redirected to the zero page (no EXEC masking in the loop).
    const long long rowBase = (long long)(n * H + hh) * W;
    const u16* bp[4];
#pragma unroll
    for (int j = 0; j < 4; ++j) {
      int ww = w0 + (j << 4) + laneRow + dw;
      const u16* p = ((unsigned)ww < (unsigned)W)
                       ? (X + (size_t)(rowBase + ww) * Kpad) : Zrow;
      bp[j] = p + laneHalf * 8;
    }

    // double-buffered K loop, last 64-chunk peeled (Kpad % 64 == 0, >= 128)
    FragU ca0, ca1, cb0, cb1, cb2, cb3;  // current
    FragU na0, na1, nb0, nb1, nb2, nb3;  // next
    LOAD_ALL(ca0, ca1, cb0, cb1, cb2, cb3, 0);

    int kb = 0;
    for (; kb < Kpad - 64; kb += 64) {
      LOAD_ALL(na0, na1, nb0, nb1, nb2, nb3, kb + 32);
      WMMA_GROUP(ca0, ca1, cb0, cb1, cb2, cb3);
      LOAD_ALL(ca0, ca1, cb0, cb1, cb2, cb3, kb + 64);
      WMMA_GROUP(na0, na1, nb0, nb1, nb2, nb3);
    }
    // tail: kb == Kpad - 64
    LOAD_ALL(na0, na1, nb0, nb1, nb2, nb3, kb + 32);
    WMMA_GROUP(ca0, ca1, cb0, cb1, cb2, cb3);
    WMMA_GROUP(na0, na1, nb0, nb1, nb2, nb3);
  }

  // D layout: lane column m = laneRow; vector elt g -> co = base + 8*laneHalf + g
#pragma unroll
  for (int i = 0; i < 2; ++i) {
    const int coS = (coT << 5) + (i << 4) + (laneHalf << 3);
#pragma unroll
    for (int j = 0; j < 4; ++j) {
      const int mS = m0 + (j << 4) + laneRow;
      float* p = Out + (size_t)mS * Copad + coS;
      *(float4*)p       = make_float4(acc[i][j][0], acc[i][j][1], acc[i][j][2], acc[i][j][3]);
      *(float4*)(p + 4) = make_float4(acc[i][j][4], acc[i][j][5], acc[i][j][6], acc[i][j][7]);
    }
  }
}

// ---------------- batch-norm statistics + apply ----------------

__global__ void k_chan_stats(const float* __restrict__ raw, int M, int C,
                             float* __restrict__ mean, float* __restrict__ inv,
                             float invM) {
  const int c = blockIdx.x;
  float s = 0.f, s2 = 0.f;
  for (int m = threadIdx.x; m < M; m += 256) {
    float v = raw[(size_t)m * C + c];
    s += v; s2 += v * v;
  }
  __shared__ float sh[256];
  __shared__ float sh2[256];
  sh[threadIdx.x] = s; sh2[threadIdx.x] = s2;
  __syncthreads();
  for (int o = 128; o > 0; o >>= 1) {
    if ((int)threadIdx.x < o) { sh[threadIdx.x] += sh[threadIdx.x + o]; sh2[threadIdx.x] += sh2[threadIdx.x + o]; }
    __syncthreads();
  }
  if (threadIdx.x == 0) {
    float mu = sh[0] * invM;
    float var = sh2[0] * invM - mu * mu;
    mean[c] = mu;
    inv[c] = rsqrtf(var + 1e-5f);
  }
}

// relu((x-mean)*inv) -> bf16, written into dst[m*dstStride + dstOff + c]
__global__ void k_bnrelu(const float* __restrict__ raw, const float* __restrict__ mean,
                         const float* __restrict__ inv, u16* __restrict__ dst,
                         int M, int rawC, int nC, int dstStride, int dstOff) {
  const long long total = (long long)M * nC;
  for (long long i = (long long)blockIdx.x * blockDim.x + threadIdx.x; i < total;
       i += (long long)gridDim.x * blockDim.x) {
    int c = (int)(i % nC);
    long long m = i / nC;
    float v = (raw[m * rawC + c] - mean[c]) * inv[c];
    dst[m * dstStride + dstOff + c] = f2bf(v > 0.f ? v : 0.f);
  }
}

// ---------------- global-pool branch ----------------

__global__ void k_pool_mean(const float* __restrict__ x, float* __restrict__ pooled,
                            int NC, int HW) {
  int i = blockIdx.x * blockDim.x + threadIdx.x;
  if (i >= NC) return;
  const float* p = x + (size_t)i * HW;
  float s = 0.f;
  for (int j = 0; j < HW; ++j) s += p[j];
  pooled[i] = s / (float)HW;
}

// pooled [8][Ci] x w [Co][Ci] -> conv, BN over batch dim, relu, bf16 vals[n][co]
__global__ void k_pool_conv_bn(const float* __restrict__ pooled, const float* __restrict__ wpool,
                               u16* __restrict__ vals, int Ci, int Co) {
  int co = blockIdx.x * blockDim.x + threadIdx.x;
  if (co >= Co) return;
  float v[8];
  float s = 0.f, s2 = 0.f;
  for (int n = 0; n < 8; ++n) {
    float a = 0.f;
    for (int ci = 0; ci < Ci; ++ci) a += pooled[n * Ci + ci] * wpool[(size_t)co * Ci + ci];
    v[n] = a; s += a; s2 += a * a;
  }
  float mu = s * 0.125f;
  float var = s2 * 0.125f - mu * mu;
  float iv = rsqrtf(var + 1e-5f);
  for (int n = 0; n < 8; ++n) {
    float t = (v[n] - mu) * iv;
    vals[n * Co + co] = f2bf(t > 0.f ? t : 0.f);
  }
}

__global__ void k_bcast_r4(const u16* __restrict__ vals, u16* __restrict__ cat,
                           int M, int HW, int C, int stride, int off) {
  const long long total = (long long)M * C;
  for (long long i = (long long)blockIdx.x * blockDim.x + threadIdx.x; i < total;
       i += (long long)gridDim.x * blockDim.x) {
    int c = (int)(i % C);
    long long m = i / C;
    int n = (int)(m / HW);
    cat[m * stride + off + c] = vals[n * C + c];
  }
}

// ---------------- PSO channel selection ----------------

__global__ void k_chan_sum_bf16(const u16* __restrict__ x, float* __restrict__ sums,
                                int M, int C) {
  const int c = blockIdx.x;
  float s = 0.f;
  for (int m = threadIdx.x; m < M; m += 256) s += bf2f(x[(size_t)m * C + c]);
  __shared__ float sh[256];
  sh[threadIdx.x] = s;
  __syncthreads();
  for (int o = 128; o > 0; o >>= 1) {
    if ((int)threadIdx.x < o) sh[threadIdx.x] += sh[threadIdx.x + o];
    __syncthreads();
  }
  if (threadIdx.x == 0) sums[c] = sh[0];
}

__global__ void k_pso(const float* __restrict__ ch_sums, const int* __restrict__ init,
                      const float* __restrict__ rands, int* __restrict__ best) {
  if (blockIdx.x != 0 || threadIdx.x != 0) return;
  const int P = 9, D = 9, IT = 100;
  float part[9][9], vel[9][9], bpos[9][9], bscore[9], gpos[9], gscore;
  for (int i = 0; i < P; ++i)
    for (int d = 0; d < D; ++d) {
      part[i][d] = (float)init[i * D + d];
      vel[i][d] = 0.f;
      bpos[i][d] = part[i][d];
    }
  for (int i = 0; i < P; ++i) {
    float s = 0.f;
    for (int d = 0; d < D; ++d) s += ch_sums[(int)floorf(part[i][d])];
    bscore[i] = s;
  }
  int gi = 0;
  for (int i = 1; i < P; ++i) if (bscore[i] < bscore[gi]) gi = i;
  gscore = bscore[gi];
  for (int d = 0; d < D; ++d) gpos[d] = part[gi][d];

  for (int it = 0; it < IT; ++it) {
    for (int i = 0; i < P; ++i) {
      const float r1 = rands[(it * P + i) * 2 + 0];
      const float r2 = rands[(it * P + i) * 2 + 1];
      float pd[9];
      float fit = 0.f;
      for (int d = 0; d < D; ++d) {
        float v = 0.7f * vel[i][d] + 1.5f * r1 * (bpos[i][d] - part[i][d])
                + 1.5f * r2 * (gpos[d] - part[i][d]);
        float np = fminf(fmaxf(part[i][d] + v, 0.f), 255.f);
        vel[i][d] = v; part[i][d] = np; pd[d] = np;
        fit += ch_sums[(int)floorf(np)];
      }
      if (fit < bscore[i]) { bscore[i] = fit; for (int d = 0; d < D; ++d) bpos[i][d] = pd[d]; }
      if (fit < gscore)    { gscore = fit;    for (int d = 0; d < D; ++d) gpos[d]  = pd[d]; }
    }
  }
  for (int d = 0; d < D; ++d) best[d] = (int)floorf(gpos[d]);
}

// x_sel conv: raw[m][co] = sum_d xm[m][best[d]] * wsel[co][d]
__global__ void k_sel_conv(const u16* __restrict__ xm, const int* __restrict__ best,
                           const float* __restrict__ wsel, float* __restrict__ raw,
                           int M, int C) {
  const long long total = (long long)M * C;
  for (long long i = (long long)blockIdx.x * blockDim.x + threadIdx.x; i < total;
       i += (long long)gridDim.x * blockDim.x) {
    int co = (int)(i % C);
    long long m = i / C;
    float a = 0.f;
    for (int d = 0; d < 9; ++d)
      a += bf2f(xm[m * 256 + best[d]]) * wsel[co * 9 + d];
    raw[i] = a;
  }
}

// ---------------- bilinear upsample + concat helpers ----------------

__global__ void k_upsample(const u16* __restrict__ src, u16* __restrict__ dst,
                           int Nb, int Hs, int Ws, int Hd, int Wd, int C,
                           int dstStride, int dstOff) {
  const long long total = (long long)Nb * Hd * Wd * C;
  const float sy = (float)Hs / (float)Hd;
  const float sx = (float)Ws / (float)Wd;
  for (long long i = (long long)blockIdx.x * blockDim.x + threadIdx.x; i < total;
       i += (long long)gridDim.x * blockDim.x) {
    int c = (int)(i % C);
    long long t = i / C;
    int wd = (int)(t % Wd); t /= Wd;
    int hd = (int)(t % Hd);
    int n = (int)(t / Hd);
    float cy = ((float)hd + 0.5f) * sy - 0.5f;
    float cx = ((float)wd + 0.5f) * sx - 0.5f;
    cy = fminf(fmaxf(cy, 0.f), (float)(Hs - 1));
    cx = fminf(fmaxf(cx, 0.f), (float)(Ws - 1));
    int y0 = (int)cy, x0 = (int)cx;
    int y1 = y0 + 1 < Hs ? y0 + 1 : Hs - 1;
    int x1 = x0 + 1 < Ws ? x0 + 1 : Ws - 1;
    float fy = cy - (float)y0, fx = cx - (float)x0;
    const size_t b = (size_t)(n * Hs) * Ws;
    float v00 = bf2f(src[(b + (size_t)y0 * Ws + x0) * C + c]);
    float v01 = bf2f(src[(b + (size_t)y0 * Ws + x1) * C + c]);
    float v10 = bf2f(src[(b + (size_t)y1 * Ws + x0) * C + c]);
    float v11 = bf2f(src[(b + (size_t)y1 * Ws + x1) * C + c]);
    float v = (1.f - fy) * ((1.f - fx) * v00 + fx * v01)
            + fy * ((1.f - fx) * v10 + fx * v11);
    size_t m = (size_t)(n * Hd + hd) * Wd + wd;
    dst[m * dstStride + dstOff + c] = f2bf(v);
  }
}

__global__ void k_fill_pad(u16* __restrict__ dst, int M, int stride, int off, int nC) {
  const long long total = (long long)M * nC;
  for (long long i = (long long)blockIdx.x * blockDim.x + threadIdx.x; i < total;
       i += (long long)gridDim.x * blockDim.x) {
    int c = (int)(i % nC);
    long long m = i / nC;
    dst[m * stride + off + c] = 0;
  }
}

// ---------------- final classifier (K=256, Co=21, fp32 NCHW out) ----------------

__global__ void k_cls2(const u16* __restrict__ hb, const float* __restrict__ w2,
                       const float* __restrict__ b2, float* __restrict__ out,
                       int M, int K, int Co, int HW) {
  const long long total = (long long)M * Co;
  for (long long i = (long long)blockIdx.x * blockDim.x + threadIdx.x; i < total;
       i += (long long)gridDim.x * blockDim.x) {
    int co = (int)(i % Co);
    long long m = i / Co;
    float a = b2[co];
    const u16* hr = hb + m * K;
    const float* wr = w2 + (size_t)co * K;
    for (int k = 0; k < K; ++k) a += bf2f(hr[k]) * wr[k];
    int n = (int)(m / HW), p = (int)(m % HW);
    out[((size_t)(n * Co + co)) * HW + p] = a;
  }
}

// ---------------- host orchestration ----------------

static inline unsigned gs_blocks(long long n) {
  long long b = (n + 255) / 256;
  if (b > 1048576) b = 1048576;
  if (b < 1) b = 1;
  return (unsigned)b;
}

extern "C" void kernel_launch(void* const* d_in, const int* in_sizes, int n_in,
                              void* d_out, int out_size, void* d_ws, size_t ws_size,
                              hipStream_t stream) {
  (void)in_sizes; (void)n_in; (void)out_size; (void)ws_size;

  const float* feat   = (const float*)d_in[0];   // (8,2048,64,64)
  const float* featlo = (const float*)d_in[1];   // (8,256,128,128)
  const float* w_low  = (const float*)d_in[2];   // (48,256,1,1)
  const float* w_a0   = (const float*)d_in[3];
  const float* w_a1   = (const float*)d_in[4];
  const float* w_a2   = (const float*)d_in[5];
  const float* w_a3   = (const float*)d_in[6];
  const float* w_pool = (const float*)d_in[7];
  const float* w_max  = (const float*)d_in[8];
  const float* w_sel  = (const float*)d_in[9];
  const float* w_proj = (const float*)d_in[10];
  const float* w_cls1 = (const float*)d_in[11];  // (256,304,3,3)
  const float* w_cls2 = (const float*)d_in[12];  // (21,256,1,1)
  const float* b_cls2 = (const float*)d_in[13];
  const float* rands  = (const float*)d_in[14];  // (100,9,2)
  const int*   initp  = (const int*)d_in[15];    // (9,9)
  float* out = (float*)d_out;

  const int M1 = 8 * 64 * 64;     // 32768
  const int M2 = 8 * 128 * 128;   // 131072

  char* base = (char*)d_ws;
  size_t cur = 0;
  auto alloc = [&](size_t bytes) -> void* {
    void* p = base + cur;
    cur += (bytes + 255) & ~(size_t)255;
    return p;
  };

  u16*   Xb    = (u16*)alloc((size_t)M1 * 2048 * 2);   // feature_out NHWC bf16
  u16*   XLb   = (u16*)alloc((size_t)M2 * 256 * 2);    // feature_low NHWC bf16
  u16*   Wa0   = (u16*)alloc((size_t)256 * 2048 * 2);
  u16*   Wa1   = (u16*)alloc((size_t)9 * 256 * 2048 * 2);
  u16*   Wa2   = (u16*)alloc((size_t)9 * 256 * 2048 * 2);
  u16*   Wa3   = (u16*)alloc((size_t)9 * 256 * 2048 * 2);
  u16*   Wmax  = (u16*)alloc((size_t)256 * 2048 * 2);
  u16*   Wproj = (u16*)alloc((size_t)256 * 1536 * 2);
  u16*   Wlow  = (u16*)alloc((size_t)64 * 256 * 2);    // Co padded 48->64
  u16*   Wcls1 = (u16*)alloc((size_t)9 * 256 * 320 * 2); // K padded 304->320
  float* raw   = (float*)alloc((size_t)M2 * 256 * 4);  // shared raw conv output
  u16*   cat1  = (u16*)alloc((size_t)M1 * 1536 * 2);
  u16*   xmb   = (u16*)alloc((size_t)M1 * 256 * 2);
  u16*   aspp  = (u16*)alloc((size_t)M1 * 256 * 2);
  u16*   cat2  = (u16*)alloc((size_t)M2 * 320 * 2);
  u16*   hbuf  = (u16*)alloc((size_t)M2 * 256 * 2);
  float* pooled  = (float*)alloc((size_t)8 * 2048 * 4);
  u16*   r4vals  = (u16*)alloc((size_t)8 * 256 * 2);
  float* meanv   = (float*)alloc(256 * 4);
  float* invv    = (float*)alloc(256 * 4);
  float* chsums  = (float*)alloc(256 * 4);
  int*   best    = (int*)alloc(16 * 4);
  u16*   zrow    = (u16*)alloc((size_t)2048 * 2);      // zero page (one K row)

  auto launch_gemm = [&](const u16* X, const u16* Wt, float* O,
                         int N, int H, int W, int K, int Co, int taps, int dil) {
    long long waves = (long long)(Co / 32) * ((long long)N * H * W / 64);
    long long threads = waves * 32;
    unsigned blocks = (unsigned)((threads + 255) / 256);
    k_gemm_wmma<<<blocks, 256, 0, stream>>>(X, Wt, zrow, O, N, H, W, K, Co, taps, dil);
  };
  auto launch_bn = [&](float* R, int M, int rawC, int nC, u16* dst, int stride, int off) {
    k_chan_stats<<<rawC, 256, 0, stream>>>(R, M, rawC, meanv, invv, 1.f / (float)M);
    k_bnrelu<<<gs_blocks((long long)M * nC), 256, 0, stream>>>(R, meanv, invv, dst, M, rawC, nC, stride, off);
  };

  // --- zero page + layout conversions ---
  k_fill_pad<<<gs_blocks(2048), 256, 0, stream>>>(zrow, 1, 0, 0, 2048);
  k_cast_nhwc<<<gs_blocks((long long)M1 * 2048), 256, 0, stream>>>(feat, Xb, 8, 2048, 4096, 2048);
  k_cast_nhwc<<<gs_blocks((long long)M2 * 256), 256, 0, stream>>>(featlo, XLb, 8, 256, 16384, 256);
  k_wcast<<<gs_blocks((long long)256 * 2048), 256, 0, stream>>>(w_a0, Wa0, 1, 256, 2048, 256, 2048);
  k_wcast<<<gs_blocks((long long)9 * 256 * 2048), 256, 0, stream>>>(w_a1, Wa1, 9, 256, 2048, 256, 2048);
  k_wcast<<<gs_blocks((long long)9 * 256 * 2048), 256, 0, stream>>>(w_a2, Wa2, 9, 256, 2048, 256, 2048);
  k_wcast<<<gs_blocks((long long)9 * 256 * 2048), 256, 0, stream>>>(w_a3, Wa3, 9, 256, 2048, 256, 2048);
  k_wcast<<<gs_blocks((long long)256 * 2048), 256, 0, stream>>>(w_max, Wmax, 1, 256, 2048, 256, 2048);
  k_wcast<<<gs_blocks((long long)256 * 1536), 256, 0, stream>>>(w_proj, Wproj, 1, 256, 1536, 256, 1536);
  k_wcast<<<gs_blocks((long long)64 * 256), 256, 0, stream>>>(w_low, Wlow, 1, 48, 256, 64, 256);
  k_wcast<<<gs_blocks((long long)9 * 256 * 320), 256, 0, stream>>>(w_cls1, Wcls1, 9, 256, 304, 256, 320);

  // --- ASPP branches r0..r3 into cat1 ---
  launch_gemm(Xb, Wa0, raw, 8, 64, 64, 2048, 256, 1, 1);
  launch_bn(raw, M1, 256, 256, cat1, 1536, 0);
  launch_gemm(Xb, Wa1, raw, 8, 64, 64, 2048, 256, 9, 12);
  launch_bn(raw, M1, 256, 256, cat1, 1536, 256);
  launch_gemm(Xb, Wa2, raw, 8, 64, 64, 2048, 256, 9, 24);
  launch_bn(raw, M1, 256, 256, cat1, 1536, 512);
  launch_gemm(Xb, Wa3, raw, 8, 64, 64, 2048, 256, 9, 36);
  launch_bn(raw, M1, 256, 256, cat1, 1536, 768);

  // --- global-pool branch r4 ---
  k_pool_mean<<<gs_blocks(8 * 2048), 256, 0, stream>>>(feat, pooled, 8 * 2048, 4096);
  k_pool_conv_bn<<<1, 256, 0, stream>>>(pooled, w_pool, r4vals, 2048, 256);
  k_bcast_r4<<<gs_blocks((long long)M1 * 256), 256, 0, stream>>>(r4vals, cat1, M1, 4096, 256, 1536, 1024);

  // --- xm branch + PSO select -> r5 ---
  launch_gemm(Xb, Wmax, raw, 8, 64, 64, 2048, 256, 1, 1);
  launch_bn(raw, M1, 256, 256, xmb, 256, 0);
  k_chan_sum_bf16<<<256, 256, 0, stream>>>(xmb, chsums, M1, 256);
  k_pso<<<1, 32, 0, stream>>>(chsums, initp, rands, best);
  k_sel_conv<<<gs_blocks((long long)M1 * 256), 256, 0, stream>>>(xmb, best, w_sel, raw, M1, 256);
  launch_bn(raw, M1, 256, 256, cat1, 1536, 1280);

  // --- projection over concat ---
  launch_gemm(cat1, Wproj, raw, 8, 64, 64, 1536, 256, 1, 1);
  launch_bn(raw, M1, 256, 256, aspp, 256, 0);

  // --- low-level branch into cat2[0..47] ---
  launch_gemm(XLb, Wlow, raw, 8, 128, 128, 256, 64, 1, 1);
  launch_bn(raw, M2, 64, 48, cat2, 320, 0);

  // --- upsample aspp into cat2[48..303], zero pad [304..319] ---
  k_upsample<<<gs_blocks((long long)M2 * 256), 256, 0, stream>>>(aspp, cat2, 8, 64, 64, 128, 128, 256, 320, 48);
  k_fill_pad<<<gs_blocks((long long)M2 * 16), 256, 0, stream>>>(cat2, M2, 320, 304, 16);

  // --- cls1 3x3 conv (K padded 320) + BN ReLU ---
  launch_gemm(cat2, Wcls1, raw, 8, 128, 128, 320, 256, 9, 1);
  launch_bn(raw, M2, 256, 256, hbuf, 256, 0);

  // --- final 1x1 conv with bias, fp32 NCHW output ---
  k_cls2<<<gs_blocks((long long)M2 * 21), 256, 0, stream>>>(hbuf, w_cls2, b_cls2, out, M2, 256, 21, 16384);
}
